// FourierSeries_28698971472231
// MI455X (gfx1250) — matrix operand order repaired and checked by reading, not compiled, not run on hardware
//
#include <hip/hip_runtime.h>

typedef __attribute__((ext_vector_type(16))) _Float16 v16h;
typedef __attribute__((ext_vector_type(8)))  _Float16 v8h;
typedef __attribute__((ext_vector_type(8)))  float    v8f;
typedef __attribute__((ext_vector_type(4)))  int      v4i;

#define D_DIM  256
#define G_NUM  5
#define K_TOT  (2 * D_DIM * G_NUM)    // 2560
#define KC     64                     // K-chunk per iteration
#define NCHUNK (K_TOT / KC)           // 40
#define CSPLIT ((D_DIM * G_NUM) / KC) // 20 cos chunks, then 20 sin chunks
#define MT     128                    // output rows per block
#define ASTR   72                     // A row stride (f16): 144B, conflict-free b128
#define BSTR   72                     // B row stride (f16)

#if __has_builtin(__builtin_amdgcn_global_load_async_to_lds_b128)
#define ASYNC_B 1
#else
#define ASYNC_B 0
#endif

__device__ __forceinline__ void wait_async_le8() {
#if __has_builtin(__builtin_amdgcn_s_wait_asynccnt)
  __builtin_amdgcn_s_wait_asynccnt(8);
#else
  asm volatile("s_wait_asynccnt 0x8" ::: "memory");
#endif
}
__device__ __forceinline__ void wait_async_0() {
#if __has_builtin(__builtin_amdgcn_s_wait_asynccnt)
  __builtin_amdgcn_s_wait_asynccnt(0);
#else
  asm volatile("s_wait_asynccnt 0x0" ::: "memory");
#endif
}

// Stage one 256x64 f16 B chunk into LDS (rows [o][k_local], stride BSTR).
__device__ __forceinline__ void stage_b(const _Float16* __restrict__ Wp,
                                        _Float16* bsbuf, int chunk, int t) {
#if ASYNC_B
  typedef __attribute__((address_space(1))) v4i gv4i;
  typedef __attribute__((address_space(3))) v4i lv4i;
  gv4i* g = (gv4i*)(Wp + (size_t)chunk * (256 * KC) + (size_t)t * KC);
  lv4i* l = (lv4i*)(bsbuf + t * BSTR);
#pragma unroll
  for (int i = 0; i < 8; ++i)
    __builtin_amdgcn_global_load_async_to_lds_b128(g + i, l + i, 0, 0);
#else
  const v8h* src = (const v8h*)(Wp + (size_t)chunk * (256 * KC) + (size_t)t * KC);
  v8h* dst = (v8h*)(bsbuf + t * BSTR);
#pragma unroll
  for (int i = 0; i < 8; ++i) dst[i] = src[i];
#endif
}

// Combine two 8xf16 LDS runs into one 16xf16 fragment register block.
__device__ __forceinline__ v16h ld_frag2(const _Float16* p, int off2) {
  v8h lo = *(const v8h*)p;
  v8h hi = *(const v8h*)(p + off2);
  return __builtin_shufflevector(lo, hi,
           0,1,2,3,4,5,6,7,8,9,10,11,12,13,14,15);
}

// ---------------------------------------------------------------------------
// Pack fourier_weight (2,256(o),256(d),5(g)) fp32 -> f16, chunk-major layout:
//   Wp[(k/64)*256*64 + o*64 + (k%64)],  k = c*1280 + d*5 + g
// ---------------------------------------------------------------------------
__global__ __launch_bounds__(256) void fourier_pack_w(
    const float* __restrict__ W, _Float16* __restrict__ Wp) {
  int e     = blockIdx.x * 256 + threadIdx.x;   // < 40*256*64 = 655360
  int kl    = e & 63;
  int o     = (e >> 6) & 255;
  int chunk = e >> 14;
  int k     = chunk * 64 + kl;
  int c     = (k >= D_DIM * G_NUM) ? 1 : 0;
  int j     = k - c * (D_DIM * G_NUM);
  Wp[e] = (_Float16)W[(size_t)(c * D_DIM + o) * (D_DIM * G_NUM) + j];
}

// ---------------------------------------------------------------------------
// Main: y = F * Wmat + b via v_wmma_f32_16x16x32_f16, f32 accumulate.
// Block = 256 threads (8 waves); block tile 128x256; wave tile 32x128.
// B tile double-buffered in LDS, filled by async global->LDS DMA one chunk
// ahead; B fragments double-buffered in registers (2 chains, unroll-2).
// ---------------------------------------------------------------------------
__global__ __launch_bounds__(256) void fourier_wmma(
    const float* __restrict__ x, const _Float16* __restrict__ Wp,
    const float* __restrict__ bias, float* __restrict__ out) {

  __shared__ __align__(16) _Float16 As[MT * ASTR];        // feature tile (18KB)
  __shared__ __align__(16) _Float16 Bs[2][256 * BSTR];    // weights 2x36KB

  const int  t    = threadIdx.x;
  const int  lane = t & 31;
  const int  wv   = t >> 5;
  const int  ln   = lane & 15;
  const bool hiH  = (lane >= 16);
  const long rowBase = (long)blockIdx.x * MT;

  const int mg = wv >> 1;            // row group 0..3  -> rows mg*32..+31
  const int ng = wv & 1;             // col group 0..1  -> cols ng*128..+127
  const int obase = ng * 128;

  v8f acc[16];                       // [0..7]: rows +0..15, [8..15]: rows +16..31
#pragma unroll
  for (int i = 0; i < 16; ++i) acc[i] = (v8f){0,0,0,0,0,0,0,0};

  // Prologue: async-stage chunk 0 into buffer 0.
  stage_b(Wp, Bs[0], 0, t);

  for (int chunk = 0; chunk < NCHUNK; ++chunk) {
    const int buf = chunk & 1;
    const int kc = chunk * KC;
    const int c  = (chunk >= CSPLIT) ? 1 : 0;        // uniform per chunk
    const int j0 = kc - c * (D_DIM * G_NUM);         // j offset in c-half
    const int d0 = j0 / G_NUM;                       // first d column used
    const int r0 = j0 - d0 * G_NUM;                  // j0 % 5

    // ---- feature tile via Chebyshev recurrence:
    //      T_{k+1}(x) = 2cos(x)*T_k(x) - T_{k-1}(x)  (cos & sin families)
#pragma unroll
    for (int i = 0; i < 8; ++i) {
      int f  = i * 256 + t;                  // 2048 (n, dl) pairs
      int n  = f >> 4;                       // row 0..127
      int dl = f & 15;                       // local d 0..15
      int d  = d0 + dl;
      if (d < D_DIM) {
        float xv   = x[(rowBase + n) * D_DIM + d];
        float c1   = __cosf(xv);
        float cur  = c ? __sinf(xv) : c1;    // T_1
        float prev = c ? 0.0f : 1.0f;        // T_0
        float twc  = 2.0f * c1;
        int   klb  = dl * G_NUM - r0;        // k_local of g=0
        _Float16* arow = &As[n * ASTR];
#pragma unroll
        for (int g = 0; g < G_NUM; ++g) {
          int kl = klb + g;
          if (kl >= 0 && kl < KC) arow[kl] = (_Float16)cur;
          float nxt = __builtin_fmaf(twc, cur, -prev);
          prev = cur; cur = nxt;
        }
      }
    }

    // ---- prefetch next chunk's B into the other buffer (async DMA) ----
    if (chunk + 1 < NCHUNK) {
      stage_b(Wp, Bs[buf ^ 1], chunk + 1, t);
      wait_async_le8();      // in-order completion: this chunk's 8 are done
    } else {
      wait_async_0();
    }
    __syncthreads();

    // ---- WMMA over this K-chunk (2 steps of K=32) ----
    const _Float16* bs = Bs[buf];
#pragma unroll
    for (int ks = 0; ks < 2; ++ks) {
      const int aoff = ks * 32 + (hiH ? 16 : 0);
      v16h aL = ld_frag2(&As[(mg * 32 +      ln) * ASTR + aoff], 8);
      v16h aH = ld_frag2(&As[(mg * 32 + 16 + ln) * ASTR + aoff], 8);
      const int boff = ks * 32 + (hiH ? 16 : 0);
      // Two independent fragment chains so loads run 2 wmma-pairs ahead.
      v16h b0 = ld_frag2(&bs[(obase +      ln) * BSTR + boff], 8);
      v16h b1 = ld_frag2(&bs[(obase + 16 + ln) * BSTR + boff], 8);
#pragma unroll
      for (int oT = 0; oT < 8; oT += 2) {
        acc[oT]       = __builtin_amdgcn_wmma_f32_16x16x32_f16(
                          false, aL, false, b0, (short)0, acc[oT],       false, false);
        acc[8 + oT]   = __builtin_amdgcn_wmma_f32_16x16x32_f16(
                          false, aH, false, b0, (short)0, acc[8 + oT],   false, false);
        if (oT + 2 < 8)
          b0 = ld_frag2(&bs[(obase + (oT + 2) * 16 + ln) * BSTR + boff], 8);
        acc[oT + 1]   = __builtin_amdgcn_wmma_f32_16x16x32_f16(
                          false, aL, false, b1, (short)0, acc[oT + 1],   false, false);
        acc[9 + oT]   = __builtin_amdgcn_wmma_f32_16x16x32_f16(
                          false, aH, false, b1, (short)0, acc[9 + oT],   false, false);
        if (oT + 3 < 8)
          b1 = ld_frag2(&bs[(obase + (oT + 3) * 16 + ln) * BSTR + boff], 8);
      }
    }
    __syncthreads();
  }

  // ---- epilogue: bias add + store (C layout: VGPR r -> M=r / M=8+r) ----
#pragma unroll
  for (int s = 0; s < 2; ++s) {
    const int mTop = mg * 32 + s * 16 + (hiH ? 8 : 0);
#pragma unroll
    for (int oT = 0; oT < 8; ++oT) {
      int col = obase + oT * 16 + ln;
      float bv = bias[col];
#pragma unroll
      for (int r = 0; r < 8; ++r) {
        long n = rowBase + mTop + r;
        out[n * D_DIM + col] = acc[s * 8 + oT][r] + bv;
      }
    }
  }
}

// ---------------------------------------------------------------------------
extern "C" void kernel_launch(void* const* d_in, const int* in_sizes, int n_in,
                              void* d_out, int out_size, void* d_ws, size_t ws_size,
                              hipStream_t stream) {
  const float* x    = (const float*)d_in[0];   // (32,2048,256) f32
  const float* W    = (const float*)d_in[1];   // (2,256,256,5) f32
  const float* bias = (const float*)d_in[2];   // (256,) f32
  float*       out  = (float*)d_out;           // (32,2048,256) f32
  _Float16*    Wp   = (_Float16*)d_ws;         // 2560*256*2 = 1.31 MB scratch

  const int nRows = in_sizes[0] / D_DIM;       // 65536

  // 1) pack + convert weights into WMMA-friendly chunked layout
  fourier_pack_w<<<(K_TOT * D_DIM) / 256, 256, 0, stream>>>(W, Wp);

  // 2) fused feature-gen (Chebyshev) + async-staged WMMA GEMM + bias
  fourier_wmma<<<nRows / MT, 256, 0, stream>>>(x, Wp, bias, out);
}
